// ClusterlingLayer_17463337026043
// MI455X (gfx1250) — compile-verified, matrix operand order checked by reading, not run on hardware
//
#include <hip/hip_runtime.h>

// Problem constants (match reference)
#define B_ROWS 131072
#define DDIM   512
#define KCL    64
#define LFIB   24
#define LCLU   64
#define VOCAB  128
#define SMOOTHF 1e-6f

typedef __attribute__((ext_vector_type(16))) __bf16 v16bf;
typedef __attribute__((ext_vector_type(8)))  float  v8f;

#define WAVES_PER_BLOCK 8
#define ROWS_PER_WAVE   16
#define ROWS_PER_BLOCK  (WAVES_PER_BLOCK * ROWS_PER_WAVE)   // 128
#define HC_STRIDE 132   // 128 + 4 pad (breaks 64-bank alignment)
#define HF_STRIDE 132

#define KCHUNKS   (DDIM / 32)        // 16
#define NFRAGS    (KCHUNKS * 4 * 32) // 2048 per-lane fragments
// workspace layout (bytes)
#define WS_WHI_OFF   0
#define WS_WLO_OFF   (NFRAGS * 16 * 2)          // 65536
#define WS_WNORM_OFF (2 * NFRAGS * 16 * 2)      // 131072

__device__ __forceinline__ v8f wmma_bf16(v16bf a, v16bf b, v8f c) {
    // (neg_a, A, neg_b, B, c_mod, C, reuse_a, reuse_b)
    return __builtin_amdgcn_wmma_f32_16x16x32_bf16(false, a, false, b, (short)0, c, false, false);
}

// Load one 16-element A fragment (K = kbase+0..7 and kbase+16..23 of a 32-chunk)
// from global f32, splitting into bf16 hi/lo parts and accumulating sum of squares.
__device__ __forceinline__ void loadFragSplit(const float* __restrict__ p,
                                              v16bf& hi, v16bf& lo, float& ssq) {
    float4 q0 = *(const float4*)(p + 0);
    float4 q1 = *(const float4*)(p + 4);
    float4 q2 = *(const float4*)(p + 16);
    float4 q3 = *(const float4*)(p + 20);
    float v[16] = {q0.x, q0.y, q0.z, q0.w, q1.x, q1.y, q1.z, q1.w,
                   q2.x, q2.y, q2.z, q2.w, q3.x, q3.y, q3.z, q3.w};
#pragma unroll
    for (int j = 0; j < 16; ++j) {
        float f = v[j];
        ssq += f * f;
        __bf16 h = (__bf16)f;
        hi[j] = h;
        lo[j] = (__bf16)(f - (float)h);
    }
}

// Same fragment pattern from LDS histograms (small integer counts: bf16 is exact).
__device__ __forceinline__ void loadFragLds(const float* p, v16bf& a) {
#pragma unroll
    for (int j = 0; j < 8; ++j) a[j] = (__bf16)p[j];
#pragma unroll
    for (int j = 0; j < 8; ++j) a[8 + j] = (__bf16)p[16 + j];
}

// ---------------------------------------------------------------------------
// Prepack: weight -> per-lane bf16 hi/lo B-fragments (exact WMMA layout) + norms.
// 2048 threads, one fragment each. Runs once per launch; ~128 KB of traffic.
// ---------------------------------------------------------------------------
__global__ __launch_bounds__(256)
void prepack_weight_kernel(const float* __restrict__ weight,
                           __bf16* __restrict__ wHi,
                           __bf16* __restrict__ wLo,
                           float*  __restrict__ wNorm) {
    const int t = blockIdx.x * blockDim.x + threadIdx.x;   // 0..2047
    const int lane = t & 31;
    const int nt   = (t >> 5) & 3;
    const int kc   = t >> 7;                               // 0..15
    const int n16  = lane & 15;
    const int kbase = (lane >> 4) * 8;

    const float* p = weight + (size_t)(nt * 16 + n16) * DDIM + kc * 32 + kbase;
    __bf16* hp = wHi + (size_t)t * 16;
    __bf16* lp = wLo + (size_t)t * 16;
#pragma unroll
    for (int j = 0; j < 16; ++j) {
        float f = p[(j < 8) ? j : (8 + j)];   // j>=8 -> p[16 + (j-8)]
        __bf16 h = (__bf16)f;
        hp[j] = h;
        lp[j] = (__bf16)(f - (float)h);
    }
    if (t < KCL) {                            // column norms, exact f32
        const float* wr = weight + (size_t)t * DDIM;
        float s = 0.f;
        for (int d = 0; d < DDIM; ++d) s += wr[d] * wr[d];
        wNorm[t] = s;
    }
}

// ---------------------------------------------------------------------------
// Main kernel
// ---------------------------------------------------------------------------
__global__ __launch_bounds__(256)
void cluster_tdist_kernel(const float* __restrict__ x,
                          const __bf16* __restrict__ wHi,
                          const __bf16* __restrict__ wLo,
                          const float*  __restrict__ wNorm,
                          const int*   __restrict__ fiber_rois,
                          const int*   __restrict__ fiber_lens,
                          const int*   __restrict__ cluster_rois,
                          const int*   __restrict__ cluster_lens,
                          float* __restrict__ out_q,
                          float* __restrict__ out_xdis) {
    __shared__ float sHistC[KCL * HC_STRIDE];                                // ~33 KB
    __shared__ float sHistF[WAVES_PER_BLOCK][ROWS_PER_WAVE * HF_STRIDE];     // ~66 KB

    const int tid  = threadIdx.x;
    const int lane = tid & 31;
    const int wave = tid >> 5;
    const int rowBase = blockIdx.x * ROWS_PER_BLOCK + wave * ROWS_PER_WAVE;

    const int n16  = lane & 15;        // N (or M) index within 16
    const int hlf  = lane >> 4;        // which half of the wave
    const int kbase = hlf * 8;         // per-lane K sub-offset inside a 32-chunk
    const int m    = n16;              // A-matrix row owned by this lane

    // ---------------- build histC (shared across block) ----------------
    for (int i = tid; i < KCL * HC_STRIDE; i += blockDim.x) sHistC[i] = 0.f;
    __syncthreads();
    if (tid < KCL) {
        int len = cluster_lens[tid];
        float* rowp = &sHistC[tid * HC_STRIDE];
        for (int i = 0; i < LCLU; ++i)
            if (i < len) rowp[cluster_rois[tid * LCLU + i]] += 1.f;
    }

    // ---------------- build histF (per wave, one lane per row) ----------------
    float* hf = &sHistF[wave][0];
    for (int i = lane; i < ROWS_PER_WAVE * HF_STRIDE; i += 32) hf[i] = 0.f;
    if (lane < ROWS_PER_WAVE) {
        int row = rowBase + lane;
        int len = fiber_lens[row];
        float* rowp = &hf[lane * HF_STRIDE];
        for (int i = 0; i < LFIB; ++i)
            if (i < len) rowp[fiber_rois[row * LFIB + i]] += 1.f;
    }
    __syncthreads();

    // ---------------- main GEMM: dot[b,k] via bf16x3 WMMA ----------------
    v8f acc[4] = {};
    float xsq = 0.f;

    const v16bf* __restrict__ whFrag = (const v16bf*)wHi;  // 32B-aligned fragments
    const v16bf* __restrict__ wlFrag = (const v16bf*)wLo;

    const float* xrow = x + (size_t)(rowBase + m) * DDIM + kbase;
    // prime the prefetch pipe for the x stream (emits global_prefetch_b8)
    __builtin_prefetch(xrow + 0 * 32, 0, 3);
    __builtin_prefetch(xrow + 1 * 32, 0, 3);

    for (int kc = 0; kc < KCHUNKS; ++kc) {
        if (kc + 2 < KCHUNKS)
            __builtin_prefetch(xrow + (kc + 2) * 32, 0, 3);
        v16bf a_hi, a_lo;
        loadFragSplit(xrow + kc * 32, a_hi, a_lo, xsq);
        const int fbase = (kc * 4) * 32 + lane;
#pragma unroll
        for (int nt = 0; nt < 4; ++nt) {
            v16bf b_hi = whFrag[fbase + nt * 32];
            v16bf b_lo = wlFrag[fbase + nt * 32];
            acc[nt] = wmma_bf16(a_hi, b_hi, acc[nt]);
            acc[nt] = wmma_bf16(a_hi, b_lo, acc[nt]);
            acc[nt] = wmma_bf16(a_lo, b_hi, acc[nt]);
        }
    }
    // each lane saw half of its x row; fold halves together
    xsq += __shfl_xor(xsq, 16, 32);

    float wsq[4];
#pragma unroll
    for (int nt = 0; nt < 4; ++nt) wsq[nt] = wNorm[nt * 16 + n16];

    // ---------------- dice GEMM: inter = histF @ histC^T (exact in bf16) ----------------
    v8f dacc[4] = {};
#pragma unroll
    for (int kc = 0; kc < VOCAB / 32; ++kc) {
        const int k0 = kc * 32;
        v16bf a;
        loadFragLds(&hf[m * HF_STRIDE + k0 + kbase], a);
#pragma unroll
        for (int nt = 0; nt < 4; ++nt) {
            v16bf b;
            loadFragLds(&sHistC[(nt * 16 + n16) * HC_STRIDE + k0 + kbase], b);
            dacc[nt] = wmma_bf16(a, b, dacc[nt]);
        }
    }

    // ---------------- epilogue ----------------
    float nF_lane = (lane < 16) ? (float)fiber_lens[rowBase + lane] : 0.f;
    float nC[4];
#pragma unroll
    for (int nt = 0; nt < 4; ++nt) nC[nt] = (float)cluster_lens[nt * 16 + n16];

    float qv[4][8];
    float rowsum[8];
#pragma unroll
    for (int r = 0; r < 8; ++r) {
        const int M = r + 8 * hlf;             // output row of this (r, lane)
        const float xn = __shfl(xsq, M, 32);
        const float nF = __shfl(nF_lane, M, 32);
        const size_t grow = (size_t)(rowBase + M) * KCL;
        float rs = 0.f;
#pragma unroll
        for (int nt = 0; nt < 4; ++nt) {
            const float dot  = acc[nt][r];
            const float xdis = xn + wsq[nt] - 2.f * dot;
            const float nCv  = nC[nt];
            float dice;
            if (nF == 0.f && nCv == 0.f)            dice = 0.f;
            else if ((nF == 0.f) != (nCv == 0.f))   dice = 1.f;
            else dice = 1.f - (2.f * dacc[nt][r] + SMOOTHF) / (nF + nCv + SMOOTHF);
            float qq = xdis * dice;
            qq = 1.f / (1.f + qq);                  // ALPHA=1 -> exponent == 1
            qv[nt][r] = qq;
            rs += qq;
            out_xdis[grow + nt * 16 + n16] = xdis;  // x_dis needs no normalization
        }
        rowsum[r] = rs;
    }
    // reduce row sums over the 16 lanes of each half (rows r and r+8 independently)
#pragma unroll
    for (int r = 0; r < 8; ++r) {
#pragma unroll
        for (int off = 1; off < 16; off <<= 1)
            rowsum[r] += __shfl_xor(rowsum[r], off, 32);
    }
#pragma unroll
    for (int r = 0; r < 8; ++r) {
        const int M = r + 8 * hlf;
        const size_t grow = (size_t)(rowBase + M) * KCL;
        const float inv = 1.f / rowsum[r];
#pragma unroll
        for (int nt = 0; nt < 4; ++nt)
            out_q[grow + nt * 16 + n16] = qv[nt][r] * inv;
    }
}

extern "C" void kernel_launch(void* const* d_in, const int* in_sizes, int n_in,
                              void* d_out, int out_size, void* d_ws, size_t ws_size,
                              hipStream_t stream) {
    (void)in_sizes; (void)n_in; (void)out_size; (void)ws_size;
    const float* x      = (const float*)d_in[0];
    const float* weight = (const float*)d_in[1];
    const int* frois    = (const int*)d_in[2];
    const int* flens    = (const int*)d_in[3];
    const int* crois    = (const int*)d_in[4];
    const int* clens    = (const int*)d_in[5];
    float* out_q    = (float*)d_out;
    float* out_xdis = out_q + (size_t)B_ROWS * KCL;

    __bf16* wHi   = (__bf16*)((char*)d_ws + WS_WHI_OFF);
    __bf16* wLo   = (__bf16*)((char*)d_ws + WS_WLO_OFF);
    float*  wNorm = (float*) ((char*)d_ws + WS_WNORM_OFF);

    // Stage 1: prepack weights into per-lane WMMA fragments (same stream -> ordered)
    hipLaunchKernelGGL(prepack_weight_kernel, dim3(NFRAGS / 256), dim3(256), 0, stream,
                       weight, wHi, wLo, wNorm);

    // Stage 2: main fused kernel
    dim3 grid(B_ROWS / ROWS_PER_BLOCK);   // 1024 blocks
    dim3 block(WAVES_PER_BLOCK * 32);     // 8 wave32s
    hipLaunchKernelGGL(cluster_tdist_kernel, grid, block, 0, stream,
                       x, wHi, wLo, wNorm, frois, flens, crois, clens, out_q, out_xdis);
}
